// Block_51556787421809
// MI455X (gfx1250) — matrix operand order repaired and checked by reading, not compiled, not run on hardware
//
#include <hip/hip_runtime.h>
#include <hip/hip_bf16.h>

// ---------------------------------------------------------------------------
// TimeSformer block for MI455X (gfx1250): bf16 WMMA GEMMs + flash attention.
// M=12544 tokens, C=768, 12 heads x 64, HID=3072. Whole working set < 192MB L2.
// GEMM: 8 waves/block share one 64-wide weight tile staged to LDS per k-step
// via ASYNCcnt-tracked global_load_async_to_lds_b128, double buffered.
// ---------------------------------------------------------------------------

typedef __attribute__((ext_vector_type(16))) __bf16 v16bf;
typedef __attribute__((ext_vector_type(8)))  float  v8f;

#define DIMC   768
#define DIM3C  2304
#define HIDD   3072
#define MTOK   12544      // 4 * 16 * 14 * 14
#define NHEAD  12
#define HD     64
#define TT_    16
#define HW_    196
#define BB_    4
#define POSPAD 208        // 196 padded to 13*16

// ---- CDNA5 async global->LDS staging (guarded: falls back to sync copy) ----
#if __has_builtin(__builtin_amdgcn_global_load_async_to_lds_b128)
#define HAVE_ASYNC_LDS 1
typedef int v4i __attribute__((__vector_size__(16)));
typedef __attribute__((address_space(1))) v4i v4i_g;   // global (__device__)
typedef __attribute__((address_space(3))) v4i v4i_l;   // LDS (__shared__)
// Generic->AS cast via integer (low 32 bits of a generic LDS address are the
// LDS byte offset on AMDGPU).
#define GPTR(p) ((v4i_g*)(size_t)(p))
#define LPTR(p) ((v4i_l*)(unsigned int)(size_t)(p))
#endif

#if __has_builtin(__builtin_amdgcn_s_wait_asynccnt)
#define WAIT_ASYNC0() __builtin_amdgcn_s_wait_asynccnt(0)
#elif defined(HAVE_ASYNC_LDS)
#define WAIT_ASYNC0() asm volatile("s_wait_asynccnt 0x0" ::: "memory")
#else
#define WAIT_ASYNC0()
#endif

__device__ __forceinline__ v8f vzero8() {
  v8f z;
#pragma unroll
  for (int i = 0; i < 8; ++i) z[i] = 0.0f;
  return z;
}

__device__ __forceinline__ v8f wmma_bf16(v16bf a, v16bf b, v8f c) {
  // D = A(16x32) * B(32x16) + C, fp32 accumulate
  return __builtin_amdgcn_wmma_f32_16x16x32_bf16(
      /*neg_a=*/false, a, /*neg_b=*/false, b,
      /*c_mod=*/(short)0, c, /*reuse_a=*/false, /*reuse_b=*/false);
}

// A-matrix 16x32 bf16 fragment from row-major [16 x ld], K-contiguous.
// Layout: lanes 0-15 = M, kh=lane>>4 selects K-halves; VGPR i:
// K = kh*16 + 2i (contiguous, matches B fragments).  kvalid pads K with 0.
__device__ __forceinline__ v16bf frag_a_rowmajor(const __bf16* p, int ld, int kvalid) {
  const int lane = threadIdx.x & 31;
  const int r = lane & 15, kh = lane >> 4;
  const __bf16* row = p + (long)r * ld;
  v16bf f;
#pragma unroll
  for (int i = 0; i < 8; ++i) {
    const int k0 = kh * 16 + 2 * i;
    f[2 * i]     = (k0     < kvalid) ? row[k0]     : (__bf16)0.0f;
    f[2 * i + 1] = (k0 + 1 < kvalid) ? row[k0 + 1] : (__bf16)0.0f;
  }
  return f;
}

// B-matrix 32x16 fragment where B[k,n] = W[n,k], W row-major [16 x ld] K-contig.
// Contiguous K order used here (matches frag_a_rowmajor, not the ISA table order): lane n = lane&15; lanes 0-15 hold K=0-15, lanes 16-31 K=16-31.
__device__ __forceinline__ v16bf frag_b_wrows(const __bf16* p, int ld) {
  const int lane = threadIdx.x & 31;
  const int n = lane & 15, kh = lane >> 4;
  const __bf16* row = p + (long)n * ld + kh * 16;
  v16bf f;
#pragma unroll
  for (int i = 0; i < 16; ++i) f[i] = row[i];
  return f;
}

// B-matrix 32x16 fragment from row-major [K x ld] with n contiguous (B[k,n]
// read directly).  Rows k >= kvalid are zero.
__device__ __forceinline__ v16bf frag_b_nmajor(const __bf16* p, int ld, int kvalid) {
  const int lane = threadIdx.x & 31;
  const int n = lane & 15, kh = lane >> 4;
  v16bf f;
#pragma unroll
  for (int i = 0; i < 8; ++i) {
    const int k0 = kh * 16 + 2 * i;
    f[2 * i]     = (k0 < kvalid)     ? p[(long)k0 * ld + n]       : (__bf16)0.0f;
    f[2 * i + 1] = (k0 + 1 < kvalid) ? p[(long)(k0 + 1) * ld + n] : (__bf16)0.0f;
  }
  return f;
}

// ---------------- reductions (wave32) ----------------
__device__ __forceinline__ float halfmax16(float x) {
#pragma unroll
  for (int m = 1; m < 16; m <<= 1) x = fmaxf(x, __shfl_xor(x, m, 32));
  return x;
}
__device__ __forceinline__ float halfsum16(float x) {
#pragma unroll
  for (int m = 1; m < 16; m <<= 1) x += __shfl_xor(x, m, 32);
  return x;
}
__device__ __forceinline__ float wavesum32(float x) {
#pragma unroll
  for (int m = 1; m < 32; m <<= 1) x += __shfl_xor(x, m, 32);
  return x;
}

// ---------------- fp32 -> bf16 convert ----------------
__global__ __launch_bounds__(256) void f2bf_kernel(const float* __restrict__ in,
                                                   __bf16* __restrict__ out, int n) {
  const int i = blockIdx.x * 256 + threadIdx.x;
  if (i < n) out[i] = (__bf16)in[i];
}

// ---------------- LayerNorm: one wave per 768-elem row, bf16 out ----------------
__global__ __launch_bounds__(256) void ln_kernel(const float* __restrict__ x,
                                                 const float* __restrict__ g,
                                                 const float* __restrict__ b,
                                                 __bf16* __restrict__ out, int rows) {
  const int wid = (blockIdx.x * blockDim.x + threadIdx.x) >> 5;
  const int lane = threadIdx.x & 31;
  if (wid >= rows) return;
  const float* row = x + (long)wid * DIMC;
  float v[24];
  float s = 0.0f;
#pragma unroll
  for (int i = 0; i < 24; ++i) { v[i] = row[lane + i * 32]; s += v[i]; }
  s = wavesum32(s);
  const float mu = s * (1.0f / DIMC);
  float vs = 0.0f;
#pragma unroll
  for (int i = 0; i < 24; ++i) { const float d = v[i] - mu; vs += d * d; }
  vs = wavesum32(vs);
  const float inv = rsqrtf(vs * (1.0f / DIMC) + 1e-5f);
  __bf16* orow = out + (long)wid * DIMC;
#pragma unroll
  for (int i = 0; i < 24; ++i) {
    const int c = lane + i * 32;
    orow[c] = (__bf16)((v[i] - mu) * inv * g[c] + b[c]);
  }
}

// Stage one 64(N) x 32(K) bf16 weight tile into LDS. 256 threads x 16B = 4KB.
__device__ __forceinline__ void stage_b_tile(__bf16* ldsDst, const __bf16* gsrc, int K) {
  const int t = threadIdx.x;
  const int n = t >> 2, q = t & 3;        // row 0..63, 16B chunk 0..3
  const __bf16* src = gsrc + (long)n * K + q * 8;
  __bf16* dst = ldsDst + n * 32 + q * 8;
#ifdef HAVE_ASYNC_LDS
  __builtin_amdgcn_global_load_async_to_lds_b128(GPTR(src), LPTR(dst), 0, 0);
#else
  *(uint4*)dst = *(const uint4*)src;
#endif
}

// ---------------- Generic GEMM: C[M,N] = A[M,K] @ W[N,K]^T ---------------------
// Block = 8 waves sharing one 64-wide N-chunk; wave w owns M-tile (mblk*8+w).
// Weight tile staged to LDS per k-step (double buffered, async prefetch).
// EPI: 0 = bias -> bf16 out         1 = bias+GELU(erf) -> bf16 out
//      2 = bias+residual -> f32 out 3 = qkv scatter temporal  4 = qkv scatter spatial
template <int EPI>
__global__ __launch_bounds__(256) void gemm_bf16(
    const __bf16* __restrict__ A, const __bf16* __restrict__ W,
    const float* __restrict__ bias, const float* __restrict__ resid,
    float* __restrict__ outF, __bf16* __restrict__ outB,
    __bf16* __restrict__ qb, __bf16* __restrict__ kb, __bf16* __restrict__ vb,
    int M, int N, int K) {
  __shared__ __align__(16) __bf16 ldsB[2][64 * 32];

  const int nchunks = N >> 6;
  const int mblk = blockIdx.x / nchunks;    // grid sized exactly: M/128 * N/64
  const int nchunk = blockIdx.x % nchunks;
  const int wslot = threadIdx.x >> 5;
  const int mtile = mblk * 8 + wslot;

  const __bf16* Ab = A + (long)mtile * 16 * K;
  const __bf16* Wb = W + (long)nchunk * 64 * K;

  v8f acc[4];
#pragma unroll
  for (int j = 0; j < 4; ++j) acc[j] = vzero8();

  const int nk = K >> 5;
  stage_b_tile(&ldsB[0][0], Wb, K);
  WAIT_ASYNC0();
  __syncthreads();

  for (int ki = 0; ki < nk; ++ki) {
    if (ki + 1 < nk) stage_b_tile(&ldsB[(ki + 1) & 1][0], Wb + (ki + 1) * 32, K);
    const v16bf va = frag_a_rowmajor(Ab + ki * 32, K, 32);
    const __bf16* bbuf = &ldsB[ki & 1][0];
#pragma unroll
    for (int j = 0; j < 4; ++j)
      acc[j] = wmma_bf16(va, frag_b_wrows(bbuf + j * 16 * 32, 32), acc[j]);
    if (ki + 1 < nk) WAIT_ASYNC0();
    __syncthreads();
  }

  const int lane = threadIdx.x & 31;
  const int nl = lane & 15, kh = lane >> 4;
#pragma unroll
  for (int j = 0; j < 4; ++j) {
    const int n = nchunk * 64 + j * 16 + nl;
#pragma unroll
    for (int vv = 0; vv < 8; ++vv) {
      const int m = mtile * 16 + vv + 8 * kh;
      const float val = acc[j][vv];
      if (EPI == 0) {
        outB[(long)m * N + n] = (__bf16)(val + bias[n]);
      } else if (EPI == 1) {
        const float t = val + bias[n];
        outB[(long)m * N + n] = (__bf16)(0.5f * t * (1.0f + erff(t * 0.70710678118654752f)));
      } else if (EPI == 2) {
        outF[(long)m * N + n] = val + bias[n] + resid[(long)m * N + n];
      } else {
        // qkv scatter (no bias in reference)
        const int which = n / DIMC;
        const int c = n - which * DIMC;
        const int head = c >> 6, d = c & 63;
        int seq, pos, posdim;
        if (EPI == 3) {  // temporal: seq over (b,hw), pos = t
          seq = m >> 4; pos = m & 15; posdim = TT_;
        } else {         // spatial: seq over (b,t), pos = hw, padded to 208
          const int bidx = m / (HW_ * TT_);
          const int r = m - bidx * (HW_ * TT_);
          pos = r >> 4;
          seq = bidx * TT_ + (r & 15);
          posdim = POSPAD;
        }
        __bf16* dst = (which == 0) ? qb : (which == 1) ? kb : vb;
        dst[(((long)seq * NHEAD + head) * posdim + pos) * HD + d] = (__bf16)val;
      }
    }
  }
}

// ---------------- Flash attention: one wave per (seq, head, q-tile) ----------
// MODE 0: temporal  (seqlen 16,  784 seqs)   MODE 1: spatial (seqlen 196, 64 seqs)
template <int MODE>
__global__ __launch_bounds__(256) void attn_kernel(
    const __bf16* __restrict__ q, const __bf16* __restrict__ k,
    const __bf16* __restrict__ v, __bf16* __restrict__ out /* [M, 768] */) {
  constexpr int SEQLEN = (MODE == 0) ? TT_ : HW_;
  constexpr int POSDIM = (MODE == 0) ? TT_ : POSPAD;
  constexpr int NSEQ   = (MODE == 0) ? BB_ * HW_ : BB_ * TT_;
  constexpr int JT     = POSDIM / 16;   // 1 or 13
  constexpr int ITILES = JT;

  __shared__ __align__(16) __bf16 ldsP[8][16 * 32];

  const int wslot = threadIdx.x >> 5;
  const int wid = blockIdx.x * 8 + wslot;
  if (wid >= NSEQ * NHEAD * ITILES) return;
  const int itile = wid % ITILES;
  const int rem = wid / ITILES;
  const int head = rem % NHEAD;
  const int seq = rem / NHEAD;

  const int lane = threadIdx.x & 31;
  const int nl = lane & 15, kh = lane >> 4;

  const long blockbase = (((long)seq * NHEAD + head) * POSDIM) * HD;
  const __bf16* qbp = q + blockbase + (long)itile * 16 * HD;
  const __bf16* kbp = k + blockbase;
  const __bf16* vbp = v + blockbase;
  __bf16* myP = &ldsP[wslot][0];

  // zero the K-pad columns (16..31) of the per-wave P tile once
#pragma unroll
  for (int vv = 0; vv < 8; ++vv) myP[(vv + 8 * kh) * 32 + 16 + nl] = (__bf16)0.0f;

  const v16bf q0 = frag_a_rowmajor(qbp, HD, 32);
  const v16bf q1 = frag_a_rowmajor(qbp + 32, HD, 32);

  v8f o[4];
#pragma unroll
  for (int c = 0; c < 4; ++c) o[c] = vzero8();
  float rm[8], rl[8];
#pragma unroll
  for (int vv = 0; vv < 8; ++vv) { rm[vv] = -3.0e38f; rl[vv] = 0.0f; }
  const float scale = 0.125f;  // 1/sqrt(64)

  for (int jt = 0; jt < JT; ++jt) {
    v8f s = vzero8();
    s = wmma_bf16(q0, frag_b_wrows(kbp + (long)jt * 16 * HD, HD), s);
    s = wmma_bf16(q1, frag_b_wrows(kbp + (long)jt * 16 * HD + 32, HD), s);

    float p[8];
#pragma unroll
    for (int vv = 0; vv < 8; ++vv) {
      float sv = s[vv] * scale;
      if (MODE == 1) { if (jt * 16 + nl >= SEQLEN) sv = -3.0e38f; }
      const float tmax = halfmax16(sv);
      const float mnew = fmaxf(rm[vv], tmax);
      const float corr = __expf(rm[vv] - mnew);
      const float pv = __expf(sv - mnew);
      rl[vv] = rl[vv] * corr + halfsum16(pv);
      rm[vv] = mnew;
      p[vv] = pv;
      o[0][vv] *= corr; o[1][vv] *= corr; o[2][vv] *= corr; o[3][vv] *= corr;
    }

    // stage P (16x16, K-padded to 32 with zeros) into the per-wave LDS tile
#pragma unroll
    for (int vv = 0; vv < 8; ++vv) myP[(vv + 8 * kh) * 32 + nl] = (__bf16)p[vv];
    asm volatile("s_wait_dscnt 0" ::: "memory");

    const v16bf pa = frag_a_rowmajor(myP, 32, 32);
#pragma unroll
    for (int c = 0; c < 4; ++c)
      o[c] = wmma_bf16(pa, frag_b_nmajor(vbp + (long)jt * 16 * HD + c * 16, HD, 16), o[c]);
  }

  // normalize and scatter back to token order [M, 768]
#pragma unroll
  for (int c = 0; c < 4; ++c) {
#pragma unroll
    for (int vv = 0; vv < 8; ++vv) {
      const float val = o[c][vv] / rl[vv];
      const int ml = vv + 8 * kh;
      const int col = head * HD + c * 16 + nl;
      long row;
      if (MODE == 0) {
        row = (long)seq * 16 + ml;
      } else {
        const int bidx = seq / TT_, t = seq % TT_;
        const int pos = itile * 16 + ml;
        if (pos >= SEQLEN) continue;
        row = (long)bidx * (HW_ * TT_) + (long)pos * TT_ + t;
      }
      out[row * DIMC + col] = (__bf16)val;
    }
  }
}

// ---------------------------------------------------------------------------
extern "C" void kernel_launch(void* const* d_in, const int* in_sizes, int n_in,
                              void* d_out, int out_size, void* d_ws, size_t ws_size,
                              hipStream_t stream) {
  (void)in_sizes; (void)n_in; (void)out_size; (void)ws_size;
  const float* x        = (const float*)d_in[0];
  const float* ln_t_g   = (const float*)d_in[1];
  const float* ln_t_b   = (const float*)d_in[2];
  const float* w_qkv_t  = (const float*)d_in[3];
  const float* w_proj_t = (const float*)d_in[4];
  const float* b_proj_t = (const float*)d_in[5];
  const float* fc_t_w   = (const float*)d_in[6];
  const float* fc_t_b   = (const float*)d_in[7];
  const float* ln1_g    = (const float*)d_in[8];
  const float* ln1_b    = (const float*)d_in[9];
  const float* w_qkv_s  = (const float*)d_in[10];
  const float* w_proj_s = (const float*)d_in[11];
  const float* b_proj_s = (const float*)d_in[12];
  const float* ln2_g    = (const float*)d_in[13];
  const float* ln2_b    = (const float*)d_in[14];
  const float* fc1_w    = (const float*)d_in[15];
  const float* fc1_b    = (const float*)d_in[16];
  const float* fc2_w    = (const float*)d_in[17];
  const float* fc2_b    = (const float*)d_in[18];
  float* outp = (float*)d_out;

  char* ws = (char*)d_ws;
  size_t off = 0;
  auto take = [&](size_t bytes) -> char* {
    char* p = ws + off; off += (bytes + 255) & ~(size_t)255; return p;
  };

  // bf16 weight copies (~20 MB)
  __bf16* wqkvT = (__bf16*)take((size_t)DIM3C * DIMC * 2);
  __bf16* wprjT = (__bf16*)take((size_t)DIMC * DIMC * 2);
  __bf16* wfcT  = (__bf16*)take((size_t)DIMC * DIMC * 2);
  __bf16* wqkvS = (__bf16*)take((size_t)DIM3C * DIMC * 2);
  __bf16* wprjS = (__bf16*)take((size_t)DIMC * DIMC * 2);
  __bf16* wfc1  = (__bf16*)take((size_t)HIDD * DIMC * 2);
  __bf16* wfc2  = (__bf16*)take((size_t)DIMC * HIDD * 2);
  // activations
  __bf16* bufA  = (__bf16*)take((size_t)MTOK * DIMC * 2);         // ln out / attn out
  const size_t qkvBytes = (size_t)(BB_ * TT_) * NHEAD * POSPAD * HD * 2;  // spatial is max
  char* regionB = take((size_t)MTOK * HIDD * 2);                  // qkv bufs / mlp hidden
  __bf16* qbuf  = (__bf16*)regionB;
  __bf16* kbuf  = (__bf16*)(regionB + qkvBytes);
  __bf16* vbuf  = (__bf16*)(regionB + 2 * qkvBytes);
  __bf16* hbuf  = (__bf16*)regionB;                               // [M, 3072] reuse
  __bf16* bufC  = (__bf16*)take((size_t)MTOK * DIMC * 2);         // proj_t out
  float*  xt    = (float*)take((size_t)MTOK * DIMC * 4);          // after temporal
  float*  x2    = (float*)take((size_t)MTOK * DIMC * 4);          // after spatial

  auto cvt = [&](const float* src, __bf16* dst, int n) {
    f2bf_kernel<<<(n + 255) / 256, 256, 0, stream>>>(src, dst, n);
  };
  cvt(w_qkv_t,  wqkvT, DIM3C * DIMC);
  cvt(w_proj_t, wprjT, DIMC * DIMC);
  cvt(fc_t_w,   wfcT,  DIMC * DIMC);
  cvt(w_qkv_s,  wqkvS, DIM3C * DIMC);
  cvt(w_proj_s, wprjS, DIMC * DIMC);
  cvt(fc1_w,    wfc1,  HIDD * DIMC);
  cvt(fc2_w,    wfc2,  DIMC * HIDD);

  const int lnBlocks = (MTOK + 7) / 8;  // 8 rows (waves) per 256-thread block
  auto gemmBlocks = [](int M, int N) { return (M / 128) * (N / 64); };  // exact

  // ---- temporal branch ----
  ln_kernel<<<lnBlocks, 256, 0, stream>>>(x, ln_t_g, ln_t_b, bufA, MTOK);
  gemm_bf16<3><<<gemmBlocks(MTOK, DIM3C), 256, 0, stream>>>(
      bufA, wqkvT, nullptr, nullptr, nullptr, nullptr, qbuf, kbuf, vbuf,
      MTOK, DIM3C, DIMC);
  attn_kernel<0><<<(BB_ * HW_ * NHEAD + 7) / 8, 256, 0, stream>>>(qbuf, kbuf, vbuf, bufA);
  gemm_bf16<0><<<gemmBlocks(MTOK, DIMC), 256, 0, stream>>>(
      bufA, wprjT, b_proj_t, nullptr, nullptr, bufC, nullptr, nullptr, nullptr,
      MTOK, DIMC, DIMC);
  gemm_bf16<2><<<gemmBlocks(MTOK, DIMC), 256, 0, stream>>>(
      bufC, wfcT, fc_t_b, x, xt, nullptr, nullptr, nullptr, nullptr,
      MTOK, DIMC, DIMC);

  // ---- spatial branch ----
  ln_kernel<<<lnBlocks, 256, 0, stream>>>(xt, ln1_g, ln1_b, bufA, MTOK);
  gemm_bf16<4><<<gemmBlocks(MTOK, DIM3C), 256, 0, stream>>>(
      bufA, wqkvS, nullptr, nullptr, nullptr, nullptr, qbuf, kbuf, vbuf,
      MTOK, DIM3C, DIMC);
  attn_kernel<1><<<(BB_ * TT_ * NHEAD * (POSPAD / 16) + 7) / 8, 256, 0, stream>>>(
      qbuf, kbuf, vbuf, bufA);
  gemm_bf16<2><<<gemmBlocks(MTOK, DIMC), 256, 0, stream>>>(
      bufA, wprjS, b_proj_s, xt, x2, nullptr, nullptr, nullptr, nullptr,
      MTOK, DIMC, DIMC);

  // ---- MLP ----
  ln_kernel<<<lnBlocks, 256, 0, stream>>>(x2, ln2_g, ln2_b, bufA, MTOK);
  gemm_bf16<1><<<gemmBlocks(MTOK, HIDD), 256, 0, stream>>>(
      bufA, wfc1, fc1_b, nullptr, nullptr, hbuf, nullptr, nullptr, nullptr,
      MTOK, HIDD, DIMC);
  gemm_bf16<2><<<gemmBlocks(MTOK, DIMC), 256, 0, stream>>>(
      hbuf, wfc2, fc2_b, x2, outp, nullptr, nullptr, nullptr, nullptr,
      MTOK, DIMC, HIDD);
}
